// DTSGNet_90082644066760
// MI455X (gfx1250) — matrix operand order repaired and checked by reading, not compiled, not run on hardware
//
#include <hip/hip_runtime.h>

#define DEVINL __device__ __forceinline__

typedef __attribute__((ext_vector_type(16))) __bf16 v16bf;
typedef __attribute__((ext_vector_type(8)))  __bf16 v8bf;
typedef __attribute__((ext_vector_type(8)))  float  v8f;

constexpr int B_  = 2048;
constexpr int L_  = 128;
constexpr int PL_ = 8;
constexpr int PS_ = 4;
constexpr int P_  = 31;           // (L-PL)/PS + 1
constexpr int H_  = 128;
constexpr int C_  = 8;
constexpr int G_  = B_ * P_;      // 63488 graphs
constexpr int ROWS_ = G_ * PL_;   // 507904 node rows

DEVINL float  bf2f(__bf16 x) { return (float)x; }
DEVINL __bf16 f2bf(float x)  { return (__bf16)x; }

// A fragment (16x32 bf16, row-major source in LDS):
// lane (l15, hi) holds row M=l15, K in {base+hi*8 .. +7} and {base+16+hi*8 .. +7}
DEVINL v16bf load_a_frag(const __bf16* row, int ks, int hi) {
  const __bf16* p = row + ks * 32 + hi * 8;
  v8bf lo = *(const v8bf*)(p);
  v8bf hh = *(const v8bf*)(p + 16);
  return __builtin_shufflevector(lo, hh, 0,1,2,3,4,5,6,7,8,9,10,11,12,13,14,15);
}
// B fragment (32x16 bf16, column-major source in LDS):
// lane (l15, hi) holds col N=l15, K in {base+hi*8 .. +7} and {base+16+hi*8 .. +7}
DEVINL v16bf load_b_frag(const __bf16* col, int ks, int hi) {
  const __bf16* p = col + ks * 32 + hi * 8;
  v8bf lo = *(const v8bf*)(p);
  v8bf hh = *(const v8bf*)(p + 16);
  return __builtin_shufflevector(lo, hh, 0,1,2,3,4,5,6,7,8,9,10,11,12,13,14,15);
}
DEVINL v8f wmma_bf16(v16bf a, v16bf b, v8f c) {
  return __builtin_amdgcn_wmma_f32_16x16x32_bf16(false, a, false, b, (short)0, c,
                                                 false, false);
}
DEVINL float sigm(float x) { return 1.f / (1.f + __expf(-x)); }

// ---------------------------------------------------------------------------
// Kernel 1: patch extraction + SAGE0 (K=2, VALU) + LayerNorm + ReLU
// block = 256 threads, 16 graphs (128 node rows) per block
// ---------------------------------------------------------------------------
__global__ void sage0_kernel(const float* __restrict__ iq,
                             const float* __restrict__ Wl0,
                             const float* __restrict__ bl0,
                             const float* __restrict__ Wr0,
                             const float* __restrict__ gamma,
                             const float* __restrict__ beta,
                             float* __restrict__ h_out) {
  extern __shared__ char smem[];
  float* sO = (float*)smem;       // 128 x 132
  float* sx = sO + 128 * 132;     // 16 graphs x 8 nodes x 2 ch
  float* mu = sx + 256;           // 128
  float* rs = mu + 128;           // 128
  const int t  = threadIdx.x;
  const int gb = blockIdx.x * 16;
  { // stage x: one element per thread; layout [gg][node][ch]
    int gg = t >> 4, n = (t >> 1) & 7, c = t & 1;
    int g = gb + gg, b = g / P_, p = g % P_;
    sx[t] = iq[(size_t)b * (2 * L_) + c * L_ + p * PS_ + n];
  }
  __syncthreads();
  for (int e = t; e < 128 * 128; e += 256) {
    int r = e >> 7, o = e & 127;
    int gg = r >> 3, i = r & 7;
    const float* xg = sx + gg * 16;
    int lo = i - 4; if (lo < 0) lo = 0;
    int hn = i + 4; if (hn > 7) hn = 7;
    float a0 = 0.f, a1 = 0.f;
    for (int j = lo; j <= hn; ++j) {
      if (j == i) continue;
      a0 += xg[j * 2 + 0]; a1 += xg[j * 2 + 1];
    }
    float inv = 1.f / (float)(hn - lo);   // neighbor count (self excluded)
    a0 *= inv; a1 *= inv;
    float x0 = xg[i * 2 + 0], x1 = xg[i * 2 + 1];
    sO[r * 132 + o] = a0 * Wl0[o * 2 + 0] + a1 * Wl0[o * 2 + 1] + bl0[o]
                    + x0 * Wr0[o * 2 + 0] + x1 * Wr0[o * 2 + 1];
  }
  __syncthreads();
  if (t < 128) {
    float s = 0.f, s2 = 0.f;
    for (int o = 0; o < 128; ++o) { float v = sO[t * 132 + o]; s += v; s2 += v * v; }
    float m = s * (1.f / 128.f);
    mu[t] = m;
    rs[t] = rsqrtf(s2 * (1.f / 128.f) - m * m + 1e-5f);
  }
  __syncthreads();
  for (int e = t; e < 128 * 128; e += 256) {
    int r = e >> 7, o = e & 127;
    float v = (sO[r * 132 + o] - mu[r]) * rs[r] * gamma[o] + beta[o];
    h_out[((size_t)gb * 8 + r) * 128 + o] = fmaxf(v, 0.f);
  }
}

// ---------------------------------------------------------------------------
// Kernel 2: SAGE layer 1/2 as WMMA GEMM:
//   out = [agg | h] (64x256) @ [Wl;Wr]^T (256x128), +bias, LN, ReLU, residual
// block = 128 threads (4 waves), 64 node rows (8 graphs) per block.
// pf_out != null -> last layer: emit mean over nodes instead of h.
// ---------------------------------------------------------------------------
__global__ void sage12_kernel(const float* __restrict__ h_in,
                              const float* __restrict__ Wl,
                              const float* __restrict__ Wr,
                              const float* __restrict__ bl,
                              const float* __restrict__ gamma,
                              const float* __restrict__ beta,
                              float* __restrict__ h_out,
                              float* __restrict__ pf_out) {
  extern __shared__ char smem[];
  __bf16* sA = (__bf16*)smem;              // 64 x 264 : [.,0..127]=agg, [.,128..255]=h
  __bf16* sW = sA + 64 * 264;              // 128 x 264 : col-major [o][k], Wl then Wr
  float*  sO = (float*)(sW + 128 * 264);   // 64 x 132
  float*  mu = sO + 64 * 132;              // 64
  float*  rs = mu + 64;                    // 64
  const int t = threadIdx.x;
  const int rowbase = blockIdx.x * 64;
  for (int e = t; e < 64 * 128; e += 128) {
    int r = e >> 7, k = e & 127;
    sA[r * 264 + 128 + k] = f2bf(h_in[((size_t)rowbase + r) * 128 + k]);
  }
  for (int e = t; e < 128 * 128; e += 128) {
    int o = e >> 7, k = e & 127;
    sW[o * 264 + k]       = f2bf(Wl[o * 128 + k]);
    sW[o * 264 + 128 + k] = f2bf(Wr[o * 128 + k]);
  }
  __syncthreads();
  // agg = A @ h (mean over in-window neighbors, per 8-node graph)
  for (int e = t; e < 64 * 128; e += 128) {
    int r = e >> 7, f = e & 127;
    int i = r & 7, gr = r & ~7;
    int lo = i - 4; if (lo < 0) lo = 0;
    int hn = i + 4; if (hn > 7) hn = 7;
    float s = 0.f;
    for (int j = lo; j <= hn; ++j) {
      if (j == i) continue;
      s += bf2f(sA[(gr + j) * 264 + 128 + f]);
    }
    sA[r * 264 + f] = f2bf(s / (float)(hn - lo));
  }
  __syncthreads();
  { // WMMA main loop: per wave 16 rows x 128 cols, K=256 (8 k-steps).
    // Batch all 8 B fragments per k-step first so the ds_loads overlap each
    // other, then fire 8 independent WMMAs back-to-back (no per-op dscnt wait).
    const int lane = t & 31, wv = t >> 5;
    const int l15 = lane & 15, hi = (lane >> 4) & 1;
    const __bf16* arow = sA + (wv * 16 + l15) * 264;
    v8f acc[8];
#pragma unroll
    for (int nt = 0; nt < 8; ++nt)
#pragma unroll
      for (int q = 0; q < 8; ++q) acc[nt][q] = 0.f;
#pragma unroll
    for (int ks = 0; ks < 8; ++ks) {
      v16bf af = load_a_frag(arow, ks, hi);
      v16bf bfr[8];
#pragma unroll
      for (int nt = 0; nt < 8; ++nt)
        bfr[nt] = load_b_frag(sW + (nt * 16 + l15) * 264, ks, hi);
#pragma unroll
      for (int nt = 0; nt < 8; ++nt)
        acc[nt] = wmma_bf16(af, bfr[nt], acc[nt]);
    }
#pragma unroll
    for (int nt = 0; nt < 8; ++nt) {
      int col = nt * 16 + l15;
      float bias = bl[col];
#pragma unroll
      for (int rr = 0; rr < 8; ++rr)
        sO[(wv * 16 + hi * 8 + rr) * 132 + col] = acc[nt][rr] + bias;
    }
  }
  __syncthreads();
  if (t < 64) {
    float s = 0.f, s2 = 0.f;
    for (int o = 0; o < 128; ++o) { float v = sO[t * 132 + o]; s += v; s2 += v * v; }
    float m = s * (1.f / 128.f);
    mu[t] = m;
    rs[t] = rsqrtf(s2 * (1.f / 128.f) - m * m + 1e-5f);
  }
  __syncthreads();
  for (int e = t; e < 64 * 128; e += 128) {
    int r = e >> 7, o = e & 127;
    float v = (sO[r * 132 + o] - mu[r]) * rs[r] * gamma[o] + beta[o];
    float hnew = h_in[((size_t)rowbase + r) * 128 + o] + fmaxf(v, 0.f);
    sO[r * 132 + o] = hnew;
    if (pf_out == nullptr)
      h_out[((size_t)rowbase + r) * 128 + o] = hnew;
  }
  if (pf_out != nullptr) {   // uniform branch
    __syncthreads();
    for (int e = t; e < 8 * 128; e += 128) {
      int gg = e >> 7, f = e & 127;
      float s = 0.f;
      for (int j = 0; j < 8; ++j) s += sO[(gg * 8 + j) * 132 + f];
      pf_out[((size_t)(rowbase >> 3) + gg) * 128 + f] = s * 0.125f;
    }
  }
}

// ---------------------------------------------------------------------------
// Kernel 3: Y(Mx384) = X(Mx128) @ W(384x128)^T + bias   (GRU input gates)
// block = 128 threads (4 waves), 64 rows per block.
// Two N-tiles per iteration: two independent accumulator chains hide both
// LDS latency and the WMMA->WMMA accumulation hazard.
// ---------------------------------------------------------------------------
__global__ void gemm_xg_kernel(const float* __restrict__ X,
                               const float* __restrict__ W,
                               const float* __restrict__ bias,
                               float* __restrict__ Y) {
  extern __shared__ char smem[];
  __bf16* sX = (__bf16*)smem;     // 64 x 136
  __bf16* sW = sX + 64 * 136;     // 384 x 136 (col-major: [n][k])
  const int t = threadIdx.x;
  const int rowbase = blockIdx.x * 64;
  for (int e = t; e < 64 * 128; e += 128) {
    int r = e >> 7, k = e & 127;
    sX[r * 136 + k] = f2bf(X[((size_t)rowbase + r) * 128 + k]);
  }
  for (int e = t; e < 384 * 128; e += 128) {
    int n = e >> 7, k = e & 127;
    sW[n * 136 + k] = f2bf(W[n * 128 + k]);
  }
  __syncthreads();
  const int lane = t & 31, wv = t >> 5;
  const int l15 = lane & 15, hi = (lane >> 4) & 1;
  const __bf16* arow = sX + (wv * 16 + l15) * 136;
  v16bf af[4];
#pragma unroll
  for (int ks = 0; ks < 4; ++ks) af[ks] = load_a_frag(arow, ks, hi);
  for (int nt2 = 0; nt2 < 12; ++nt2) {
    const int nt0 = nt2 * 2, nt1 = nt0 + 1;
    v8f acc0, acc1;
#pragma unroll
    for (int q = 0; q < 8; ++q) { acc0[q] = 0.f; acc1[q] = 0.f; }
    const __bf16* bcol0 = sW + (nt0 * 16 + l15) * 136;
    const __bf16* bcol1 = sW + (nt1 * 16 + l15) * 136;
    v16bf b0[4], b1[4];
#pragma unroll
    for (int ks = 0; ks < 4; ++ks) {
      b0[ks] = load_b_frag(bcol0, ks, hi);
      b1[ks] = load_b_frag(bcol1, ks, hi);
    }
#pragma unroll
    for (int ks = 0; ks < 4; ++ks) {
      acc0 = wmma_bf16(af[ks], b0[ks], acc0);
      acc1 = wmma_bf16(af[ks], b1[ks], acc1);
    }
    const int col0 = nt0 * 16 + l15, col1 = nt1 * 16 + l15;
    const float bv0 = bias[col0], bv1 = bias[col1];
#pragma unroll
    for (int rr = 0; rr < 8; ++rr) {
      size_t m = (size_t)rowbase + wv * 16 + hi * 8 + rr;
      Y[m * 384 + col0] = acc0[rr] + bv0;
      Y[m * 384 + col1] = acc1[rr] + bv1;
    }
  }
}

// ---------------------------------------------------------------------------
// Kernel 4: persistent GRU recurrence. Each block owns 32 batch rows and runs
// all 31 time steps with h (f32+bf16) and w_hh (bf16) resident in LDS.
// block = 256 threads (8 waves): wave -> (row-tile of 16, 96-col group).
// ---------------------------------------------------------------------------
__global__ void gru_rec_kernel(const float* __restrict__ xg,
                               const float* __restrict__ w_hh,
                               const float* __restrict__ b_hh,
                               float* __restrict__ out_seq,
                               float* __restrict__ g_out) {
  extern __shared__ char smem[];
  __bf16* sW = (__bf16*)smem;             // 384 x 136 col-major
  __bf16* hb = sW + 384 * 136;            // 32 x 136 (bf16 h for WMMA A)
  float*  hf = (float*)(hb + 32 * 136);   // 32 x 132 (f32 master h)
  float*  hg = hf + 32 * 132;             // 32 x 388 (gate preacts stage)
  const int t = threadIdx.x;
  const int bbase = blockIdx.x * 32;
  for (int e = t; e < 384 * 128; e += 256) {
    int n = e >> 7, k = e & 127;
    sW[n * 136 + k] = f2bf(w_hh[n * 128 + k]);
  }
  for (int e = t; e < 32 * 136; e += 256) hb[e] = f2bf(0.f);
  for (int e = t; e < 32 * 132; e += 256) hf[e] = 0.f;
  __syncthreads();
  const int lane = t & 31, wv = t >> 5;
  const int l15 = lane & 15, hi = (lane >> 4) & 1;
  const int rt = wv & 1, ng = wv >> 1;
  const __bf16* arow = hb + (rt * 16 + l15) * 136;
  for (int tt = 0; tt < P_; ++tt) {
    { // hg = h @ w_hh^T  (16 rows x 96 cols per wave, K=128)
      v16bf af[4];
#pragma unroll
      for (int ks = 0; ks < 4; ++ks) af[ks] = load_a_frag(arow, ks, hi);
#pragma unroll
      for (int jj = 0; jj < 3; ++jj) {
        const int nt0 = ng * 6 + jj * 2, nt1 = nt0 + 1;
        v8f acc0, acc1;
#pragma unroll
        for (int q = 0; q < 8; ++q) { acc0[q] = 0.f; acc1[q] = 0.f; }
        const __bf16* bcol0 = sW + (nt0 * 16 + l15) * 136;
        const __bf16* bcol1 = sW + (nt1 * 16 + l15) * 136;
        v16bf b0[4], b1[4];
#pragma unroll
        for (int ks = 0; ks < 4; ++ks) {
          b0[ks] = load_b_frag(bcol0, ks, hi);
          b1[ks] = load_b_frag(bcol1, ks, hi);
        }
#pragma unroll
        for (int ks = 0; ks < 4; ++ks) {
          acc0 = wmma_bf16(af[ks], b0[ks], acc0);
          acc1 = wmma_bf16(af[ks], b1[ks], acc1);
        }
        const int col0 = nt0 * 16 + l15, col1 = nt1 * 16 + l15;
#pragma unroll
        for (int rr = 0; rr < 8; ++rr) {
          hg[(rt * 16 + hi * 8 + rr) * 388 + col0] = acc0[rr];
          hg[(rt * 16 + hi * 8 + rr) * 388 + col1] = acc1[rr];
        }
      }
    }
    __syncthreads();
    for (int e = t; e < 32 * 128; e += 256) {
      int lr = e >> 7, j = e & 127;
      int b = bbase + lr;
      const float* xr = xg + ((size_t)b * P_ + tt) * 384;
      float r  = sigm(xr[j]       + hg[lr * 388 + j]       + b_hh[j]);
      float z  = sigm(xr[128 + j] + hg[lr * 388 + 128 + j] + b_hh[128 + j]);
      float hgn = hg[lr * 388 + 256 + j] + b_hh[256 + j];
      float nn = tanhf(xr[256 + j] + r * hgn);
      float hp = hf[lr * 132 + j];
      float hnew = (1.f - z) * nn + z * hp;
      hf[lr * 132 + j] = hnew;
      hb[lr * 136 + j] = f2bf(hnew);
      if (out_seq) out_seq[((size_t)b * P_ + tt) * 128 + j] = hnew;
      if (g_out && tt == P_ - 1) g_out[(size_t)b * 128 + j] = hnew;
    }
    __syncthreads();
  }
}

// ---------------------------------------------------------------------------
// Kernel 5: logits = g (2048x128) @ etf (128x8)   (tiny)
// ---------------------------------------------------------------------------
__global__ void logits_kernel(const float* __restrict__ g,
                              const float* __restrict__ etf,
                              float* __restrict__ logits) {
  int idx = blockIdx.x * blockDim.x + threadIdx.x;   // B_*C_ threads
  int b = idx >> 3, c = idx & 7;
  float s = 0.f;
  for (int h = 0; h < H_; ++h) s += g[(size_t)b * H_ + h] * etf[h * C_ + c];
  logits[idx] = s;
}

// ---------------------------------------------------------------------------
extern "C" void kernel_launch(void* const* d_in, const int* in_sizes, int n_in,
                              void* d_out, int out_size, void* d_ws, size_t ws_size,
                              hipStream_t stream) {
  (void)in_sizes; (void)n_in; (void)out_size; (void)ws_size;
  const float* iq    = (const float*)d_in[0];
  const float* Wl0   = (const float*)d_in[1];
  const float* bl0   = (const float*)d_in[2];
  const float* Wr0   = (const float*)d_in[3];
  const float* Wl12  = (const float*)d_in[4];   // (2,H,H)
  const float* bl12  = (const float*)d_in[5];   // (2,H)
  const float* Wr12  = (const float*)d_in[6];   // (2,H,H)
  const float* ln_g  = (const float*)d_in[7];   // (3,H)
  const float* ln_b  = (const float*)d_in[8];   // (3,H)
  const float* gw_ih = (const float*)d_in[9];   // (2,3H,H)
  const float* gw_hh = (const float*)d_in[10];  // (2,3H,H)
  const float* gb_ih = (const float*)d_in[11];  // (2,3H)
  const float* gb_hh = (const float*)d_in[12];  // (2,3H)
  const float* etf   = (const float*)d_in[13];  // (H,C)

  float* logits = (float*)d_out;                 // B*C
  float* g      = logits + (size_t)B_ * C_;      // B*H
  float* pf     = g + (size_t)B_ * H_;           // G*H (patch_features)

  float* h_buf = (float*)d_ws;                   // ROWS_*H  (260 MB)
  float* xg    = h_buf + (size_t)ROWS_ * H_;     // G*3H    (97.5 MB)
  float* seq1  = xg + (size_t)G_ * 3 * H_;       // G*H     (32.5 MB)

  const size_t smem0 = (size_t)(128 * 132 + 256 + 128 + 128) * 4;
  const size_t smemS = (size_t)(64 * 264 + 128 * 264) * 2 +
                       (size_t)(64 * 132 + 128) * 4;
  const size_t smemG = (size_t)(64 * 136 + 384 * 136) * 2;
  const size_t smemR = (size_t)(384 * 136 + 32 * 136) * 2 +
                       (size_t)(32 * 132 + 32 * 388) * 4;

  sage0_kernel<<<G_ / 16, 256, smem0, stream>>>(iq, Wl0, bl0, Wr0, ln_g, ln_b, h_buf);

  sage12_kernel<<<ROWS_ / 64, 128, smemS, stream>>>(
      h_buf, Wl12, Wr12, bl12, ln_g + H_, ln_b + H_, h_buf, nullptr);
  sage12_kernel<<<ROWS_ / 64, 128, smemS, stream>>>(
      h_buf, Wl12 + H_ * H_, Wr12 + H_ * H_, bl12 + H_,
      ln_g + 2 * H_, ln_b + 2 * H_, h_buf, pf);

  // GRU layer 1
  gemm_xg_kernel<<<G_ / 64, 128, smemG, stream>>>(pf, gw_ih, gb_ih, xg);
  gru_rec_kernel<<<B_ / 32, 256, smemR, stream>>>(xg, gw_hh, gb_hh, seq1, nullptr);
  // GRU layer 2
  gemm_xg_kernel<<<G_ / 64, 128, smemG, stream>>>(
      seq1, gw_ih + 3 * H_ * H_, gb_ih + 3 * H_, xg);
  gru_rec_kernel<<<B_ / 32, 256, smemR, stream>>>(
      xg, gw_hh + 3 * H_ * H_, gb_hh + 3 * H_, nullptr, g);

  logits_kernel<<<(B_ * C_) / 256, 256, 0, stream>>>(g, etf, logits);
}
